// KiperwasserDependencyParser_49993419325441
// MI455X (gfx1250) — compile-verified, matrix operand order checked by reading, not run on hardware
//
#include <hip/hip_runtime.h>

// ---------------------------------------------------------------------------
// Types for CDNA5 WMMA (wave32, v_wmma_f32_16x16x32_bf16)
// ---------------------------------------------------------------------------
typedef __attribute__((ext_vector_type(16))) __bf16     v16bf;
typedef __attribute__((ext_vector_type(8)))  float      v8f;
typedef __attribute__((ext_vector_type(4)))  unsigned   u32x4;
typedef unsigned short ushort_t;

union FragBF {
    u32x4 q[2];   // two 16-byte chunks
    v16bf v;      // 16 bf16 = 8 VGPRs
};

// Model constants
enum { N_TOK = 1024, DW = 300, DP = 100, DIN = 400, DIN_P = 416,
       H = 512, G4H = 2048, TWOH = 1024, MDIM = 256 };

__device__ __forceinline__ ushort_t f32_to_bf16_rne(float f) {
    unsigned u = __float_as_uint(f);
    unsigned r = (u + 0x7FFFu + ((u >> 16) & 1u)) >> 16;
    return (ushort_t)r;
}

__device__ __forceinline__ float sigm(float x) {
    return 1.0f / (1.0f + __expf(-x));
}

// ---------------------------------------------------------------------------
// Embed words+pos, concat, pad K 400->416, emit bf16 [N_TOK, 416]
// ---------------------------------------------------------------------------
__global__ void embed_concat_bf16(const int* __restrict__ widx,
                                  const int* __restrict__ pidx,
                                  const float* __restrict__ wemb,
                                  const float* __restrict__ pemb,
                                  ushort_t* __restrict__ xbf) {
    int idx = blockIdx.x * blockDim.x + threadIdx.x;
    if (idx >= N_TOK * DIN_P) return;
    int i = idx / DIN_P, k = idx - i * DIN_P;
    float v = 0.0f;
    if (k < DW)       v = wemb[widx[i] * DW + k];
    else if (k < DIN) v = pemb[pidx[i] * DP + (k - DW)];
    xbf[idx] = f32_to_bf16_rne(v);
}

// ---------------------------------------------------------------------------
// Generic f32 -> bf16 weight repack with column offset + K zero-pad
// dst[r*Kp + k] = (k < Kcopy) ? bf16(src[r*srcStride + colOff + k]) : 0
// ---------------------------------------------------------------------------
__global__ void conv_bf16(const float* __restrict__ src, ushort_t* __restrict__ dst,
                          int rows, int srcStride, int colOff, int Kcopy, int Kp) {
    int idx = blockIdx.x * blockDim.x + threadIdx.x;
    if (idx >= rows * Kp) return;
    int r = idx / Kp, k = idx - r * Kp;
    float f = (k < Kcopy) ? src[r * srcStride + colOff + k] : 0.0f;
    dst[idx] = f32_to_bf16_rne(f);
}

// ---------------------------------------------------------------------------
// Transpose f32 [R,C] -> [C,R] (for coalesced recurrent matvec)
// ---------------------------------------------------------------------------
__global__ void transpose_f32(const float* __restrict__ src, float* __restrict__ dst,
                              int R, int C) {
    int idx = blockIdx.x * blockDim.x + threadIdx.x;
    if (idx >= R * C) return;
    int r = idx / C, c = idx - r * C;
    dst[c * R + r] = src[idx];
}

// ---------------------------------------------------------------------------
// WMMA GEMM: C[M,N] = A[M,K](bf16) * W[N,K]^T(bf16) + bias[N]
// One 16x16 C tile per wave, 8 waves/block. K multiple of 32.
// Fragment layout per CDNA5 ISA 7.12.2:
//   A (16-bit, 16x32): lane m(=lane&15), half=lane>>4 holds
//     elems 0..7  = A[m][k0 + 8*half .. +7]
//     elems 8..15 = A[m][k0 + 16 + 8*half .. +7]
//   B operand loaded as W rows (B = W^T, so B-fragment == A-layout of W)
//   C/D: row = v + 8*half, col = lane&15
// ---------------------------------------------------------------------------
__global__ __launch_bounds__(256)
void wmma_gemm_bias(const ushort_t* __restrict__ A, const ushort_t* __restrict__ W,
                    const float* __restrict__ bias, float* __restrict__ C,
                    int M, int N, int K) {
    int lane = threadIdx.x & 31;
    int wave = threadIdx.x >> 5;
    int tilesN = N >> 4;
    int tile  = blockIdx.x * 8 + wave;
    int tM = tile / tilesN, tN = tile - tM * tilesN;
    int half = lane >> 4, l16 = lane & 15;

    const ushort_t* pa = A + (size_t)(tM * 16 + l16) * K;
    const ushort_t* pw = W + (size_t)(tN * 16 + l16) * K;

    float bv = bias ? bias[tN * 16 + l16] : 0.0f;
    v8f acc = { bv, bv, bv, bv, bv, bv, bv, bv };

    int koff = 8 * half;
    for (int k0 = 0; k0 < K; k0 += 32) {
        FragBF fa, fb;
        fa.q[0] = *(const u32x4*)(pa + k0 + koff);
        fa.q[1] = *(const u32x4*)(pa + k0 + 16 + koff);
        fb.q[0] = *(const u32x4*)(pw + k0 + koff);
        fb.q[1] = *(const u32x4*)(pw + k0 + 16 + koff);
        acc = __builtin_amdgcn_wmma_f32_16x16x32_bf16(
            false, fa.v, false, fb.v, (short)0, acc, false, false);
    }

    float* pc = C + (size_t)(tM * 16 + 8 * half) * N + tN * 16 + l16;
#pragma unroll
    for (int v = 0; v < 8; ++v) pc[(size_t)v * N] = acc[v];
}

// ---------------------------------------------------------------------------
// One bidirectional LSTM layer. grid=2 blocks (dir 0=fwd, 1=rev), 1024 threads.
// zPre = X@Wih^T + b precomputed [N_TOK, 2048]. whhT is Whh transposed [512,2048]
// so lane g reads column g of row k (coalesced). h,c state in LDS.
// Writes h (f32) and bf16 copy into its 512-column half of [N_TOK, 1024].
// ---------------------------------------------------------------------------
__global__ __launch_bounds__(1024)
void lstm_layer(const float* __restrict__ zF, const float* __restrict__ zR,
                const float* __restrict__ whhTF, const float* __restrict__ whhTR,
                float* __restrict__ hOut, ushort_t* __restrict__ hOutBf, int n) {
    __shared__ float hCur[H];
    __shared__ float cSt[H];
    __shared__ float zb[G4H];

    int tid = threadIdx.x;
    int dir = blockIdx.x;
    const float* zPre = dir ? zR : zF;
    const float* whhT = dir ? whhTR : whhTF;
    int colOff = dir ? H : 0;

    if (tid < H) { hCur[tid] = 0.0f; cSt[tid] = 0.0f; }
    __syncthreads();

    for (int s = 0; s < n; ++s) {
        int t = dir ? (n - 1 - s) : s;
        const float* zrow = zPre + (size_t)t * G4H;
        // z = zPre + h @ Whh^T  (each thread: 2 outputs of 512 MACs)
        for (int g = tid; g < G4H; g += 1024) {
            float acc = zrow[g];
            const float* wp = whhT + g;
#pragma unroll 8
            for (int k = 0; k < H; ++k) acc += wp[(size_t)k * G4H] * hCur[k];
            zb[g] = acc;
        }
        __syncthreads();
        if (tid < H) {
            float zi = zb[tid], zf = zb[H + tid], zg = zb[2 * H + tid], zo = zb[3 * H + tid];
            float cc = sigm(zf) * cSt[tid] + sigm(zi) * tanhf(zg);
            float hh = sigm(zo) * tanhf(cc);
            cSt[tid] = cc;
            hCur[tid] = hh;
            size_t o = (size_t)t * TWOH + colOff + tid;
            hOut[o]   = hh;
            hOutBf[o] = f32_to_bf16_rne(hh);
        }
        __syncthreads();
    }
}

// ---------------------------------------------------------------------------
// Pairwise scores: out[i][j] = b2 + sum_m tanh(a[i][m] + bp[j][m]) * w2[m]
// 32x32 tile per block, m reduced in 2 chunks of 128 (LDS ~34KB).
// sb padded to stride 129 -> conflict-free across j lanes (129 % 64 == 1).
// ---------------------------------------------------------------------------
__global__ __launch_bounds__(256)
void pairwise_scores(const float* __restrict__ a, const float* __restrict__ bp,
                     const float* __restrict__ w2, const float* __restrict__ b2,
                     float* __restrict__ out) {
    __shared__ float sa[32 * 128];
    __shared__ float sb[32 * 129];
    __shared__ float w2s[MDIM];

    int tid = threadIdx.x;
    int i0 = blockIdx.y * 32, j0 = blockIdx.x * 32;
    w2s[tid] = w2[tid];

    int j = tid & 31, ib = tid >> 5;      // per-wave: same ib, j = lane
    float accv[4] = {0.f, 0.f, 0.f, 0.f};

    for (int c0 = 0; c0 < MDIM; c0 += 128) {
        __syncthreads();
        for (int idx = tid; idx < 32 * 128; idx += 256) {
            int r = idx >> 7, c = idx & 127;
            sa[idx]          = a [(size_t)(i0 + r) * MDIM + c0 + c];
            sb[r * 129 + c]  = bp[(size_t)(j0 + r) * MDIM + c0 + c];
        }
        __syncthreads();
#pragma unroll
        for (int s = 0; s < 4; ++s) {
            int i = ib * 4 + s;
            const float* pa = sa + i * 128;
            const float* pb = sb + j * 129;
            float sum = 0.0f;
            for (int c = 0; c < 128; ++c)
                sum += tanhf(pa[c] + pb[c]) * w2s[c0 + c];
            accv[s] += sum;
        }
    }
    float bb = b2[0];
#pragma unroll
    for (int s = 0; s < 4; ++s) {
        int i = ib * 4 + s;
        out[(size_t)(i0 + i) * N_TOK + j0 + j] = accv[s] + bb;
    }
}

// ---------------------------------------------------------------------------
// Launch
// ---------------------------------------------------------------------------
extern "C" void kernel_launch(void* const* d_in, const int* in_sizes, int n_in,
                              void* d_out, int out_size, void* d_ws, size_t ws_size,
                              hipStream_t stream) {
    (void)in_sizes; (void)n_in; (void)out_size; (void)ws_size;

    const int*   word_idx = (const int*)  d_in[0];
    const int*   pos_idx  = (const int*)  d_in[1];
    const float* word_emb = (const float*)d_in[2];
    const float* pos_emb  = (const float*)d_in[3];
    const float* wih0f = (const float*)d_in[4];
    const float* whh0f = (const float*)d_in[5];
    const float* b0f   = (const float*)d_in[6];
    const float* wih0r = (const float*)d_in[7];
    const float* whh0r = (const float*)d_in[8];
    const float* b0r   = (const float*)d_in[9];
    const float* wih1f = (const float*)d_in[10];
    const float* whh1f = (const float*)d_in[11];
    const float* b1f   = (const float*)d_in[12];
    const float* wih1r = (const float*)d_in[13];
    const float* whh1r = (const float*)d_in[14];
    const float* b1r   = (const float*)d_in[15];
    const float* W1    = (const float*)d_in[16];
    const float* bmlp1 = (const float*)d_in[17];
    const float* w2    = (const float*)d_in[18];
    const float* b2    = (const float*)d_in[19];
    float* out = (float*)d_out;

    // ---- workspace layout (256B aligned) ----
    char* base = (char*)d_ws;
    size_t off = 0;
    auto alloc = [&](size_t bytes) -> char* {
        char* p = base + off;
        off += (bytes + 255) & ~(size_t)255;
        return p;
    };
    ushort_t* xbf    = (ushort_t*)alloc((size_t)N_TOK * DIN_P * 2);
    ushort_t* wih0fb = (ushort_t*)alloc((size_t)G4H * DIN_P * 2);
    ushort_t* wih0rb = (ushort_t*)alloc((size_t)G4H * DIN_P * 2);
    ushort_t* wih1fb = (ushort_t*)alloc((size_t)G4H * TWOH * 2);
    ushort_t* wih1rb = (ushort_t*)alloc((size_t)G4H * TWOH * 2);
    ushort_t* w1hb   = (ushort_t*)alloc((size_t)MDIM * TWOH * 2);
    ushort_t* w1db   = (ushort_t*)alloc((size_t)MDIM * TWOH * 2);
    float* whh0fT = (float*)alloc((size_t)H * G4H * 4);
    float* whh0rT = (float*)alloc((size_t)H * G4H * 4);
    float* whh1fT = (float*)alloc((size_t)H * G4H * 4);
    float* whh1rT = (float*)alloc((size_t)H * G4H * 4);
    float* Zf     = (float*)alloc((size_t)N_TOK * G4H * 4);   // reused layer0/1
    float* Zr     = (float*)alloc((size_t)N_TOK * G4H * 4);
    float* h0     = (float*)alloc((size_t)N_TOK * TWOH * 4);
    ushort_t* h0b = (ushort_t*)alloc((size_t)N_TOK * TWOH * 2);
    float* h1     = (float*)alloc((size_t)N_TOK * TWOH * 4);
    ushort_t* h1b = (ushort_t*)alloc((size_t)N_TOK * TWOH * 2);
    float* ap     = (float*)alloc((size_t)N_TOK * MDIM * 4);
    float* bpj    = (float*)alloc((size_t)N_TOK * MDIM * 4);
    (void)h0;

    auto cdiv = [](int a, int b) { return (a + b - 1) / b; };

    // 1) embed + concat -> bf16 padded
    embed_concat_bf16<<<cdiv(N_TOK * DIN_P, 256), 256, 0, stream>>>(
        word_idx, pos_idx, word_emb, pos_emb, xbf);

    // 2) repack weights to bf16 (K-padded)
    conv_bf16<<<cdiv(G4H * DIN_P, 256), 256, 0, stream>>>(wih0f, wih0fb, G4H, DIN, 0, DIN, DIN_P);
    conv_bf16<<<cdiv(G4H * DIN_P, 256), 256, 0, stream>>>(wih0r, wih0rb, G4H, DIN, 0, DIN, DIN_P);
    conv_bf16<<<cdiv(G4H * TWOH, 256), 256, 0, stream>>>(wih1f, wih1fb, G4H, TWOH, 0, TWOH, TWOH);
    conv_bf16<<<cdiv(G4H * TWOH, 256), 256, 0, stream>>>(wih1r, wih1rb, G4H, TWOH, 0, TWOH, TWOH);
    conv_bf16<<<cdiv(MDIM * TWOH, 256), 256, 0, stream>>>(W1, w1hb, MDIM, 2 * TWOH, 0,    TWOH, TWOH);
    conv_bf16<<<cdiv(MDIM * TWOH, 256), 256, 0, stream>>>(W1, w1db, MDIM, 2 * TWOH, TWOH, TWOH, TWOH);

    // 3) transpose recurrent weights for coalesced matvec
    transpose_f32<<<cdiv(G4H * H, 256), 256, 0, stream>>>(whh0f, whh0fT, G4H, H);
    transpose_f32<<<cdiv(G4H * H, 256), 256, 0, stream>>>(whh0r, whh0rT, G4H, H);
    transpose_f32<<<cdiv(G4H * H, 256), 256, 0, stream>>>(whh1f, whh1fT, G4H, H);
    transpose_f32<<<cdiv(G4H * H, 256), 256, 0, stream>>>(whh1r, whh1rT, G4H, H);

    // 4) layer-0 input projections: Z = X @ Wih^T + b   [1024, 2048]
    int blkZ0 = (N_TOK / 16) * (G4H / 16) / 8;   // 1024
    wmma_gemm_bias<<<blkZ0, 256, 0, stream>>>(xbf, wih0fb, b0f, Zf, N_TOK, G4H, DIN_P);
    wmma_gemm_bias<<<blkZ0, 256, 0, stream>>>(xbf, wih0rb, b0r, Zr, N_TOK, G4H, DIN_P);

    // 5) layer-0 recurrence (both directions concurrently)
    lstm_layer<<<2, 1024, 0, stream>>>(Zf, Zr, whh0fT, whh0rT, h0, h0b, N_TOK);

    // 6) layer-1 input projections
    wmma_gemm_bias<<<blkZ0, 256, 0, stream>>>(h0b, wih1fb, b1f, Zf, N_TOK, G4H, TWOH);
    wmma_gemm_bias<<<blkZ0, 256, 0, stream>>>(h0b, wih1rb, b1r, Zr, N_TOK, G4H, TWOH);

    // 7) layer-1 recurrence
    lstm_layer<<<2, 1024, 0, stream>>>(Zf, Zr, whh1fT, whh1rT, h1, h1b, N_TOK);

    // 8) MLP head/dep projections: a = h1@W1h^T + bmlp1 ; bp = h1@W1d^T
    int blkP = (N_TOK / 16) * (MDIM / 16) / 8;   // 128
    wmma_gemm_bias<<<blkP, 256, 0, stream>>>(h1b, w1hb, bmlp1,  ap,  N_TOK, MDIM, TWOH);
    wmma_gemm_bias<<<blkP, 256, 0, stream>>>(h1b, w1db, nullptr, bpj, N_TOK, MDIM, TWOH);

    // 9) pairwise tanh reduction -> scores [1024, 1024]
    dim3 grid(N_TOK / 32, N_TOK / 32);
    pairwise_scores<<<grid, 256, 0, stream>>>(ap, bpj, w2, b2, out);
}
